// ClusterFusion_73813307949072
// MI455X (gfx1250) — matrix operand order repaired and checked by reading, not compiled, not run on hardware
//
#include <hip/hip_runtime.h>
#include <math.h>

#define B_  16
#define T_  750
#define F_  2048
#define C_  20
#define TF_ 1536000        // T_*F_
#define TC_ 15000          // T_*C_
#define KC  512            // f-chunk staged per step
#define FUSED_N 24576000   // B_*T_*F_   (x elements)
#define CENT_N  30720000   // C_*T_*F_   (center elements)

typedef __bf16 v16bf __attribute__((ext_vector_type(16)));
typedef __bf16 v8bf  __attribute__((ext_vector_type(8)));
typedef __bf16 v4bf  __attribute__((ext_vector_type(4)));
typedef float  v8f   __attribute__((ext_vector_type(8)));
typedef float  v4f   __attribute__((ext_vector_type(4)));

// ---------------------------------------------------------------------------
// Kernel 0: per-row sum of squares (fp32).  blocks 0..15 -> xsq, 16..35 -> csq
// ---------------------------------------------------------------------------
__global__ __launch_bounds__(256) void sumsq_kernel(const float* __restrict__ x,
                                                    const float* __restrict__ cent,
                                                    float* __restrict__ ws) {
    const int row = blockIdx.x;
    const float* p = (row < B_) ? (x + (size_t)row * TF_)
                                : (cent + (size_t)(row - B_) * TF_);
    float s = 0.0f;
    for (int i = threadIdx.x; i < TF_ / 4; i += 256) {
        v4f v = *(const v4f*)&p[4 * i];
        s += v[0] * v[0] + v[1] * v[1] + v[2] * v[2] + v[3] * v[3];
    }
    __shared__ float red[256];
    red[threadIdx.x] = s;
    __syncthreads();
    for (int off = 128; off > 0; off >>= 1) {
        if (threadIdx.x < off) red[threadIdx.x] += red[threadIdx.x + off];
        __syncthreads();
    }
    if (threadIdx.x == 0) ws[row] = red[0];   // ws[0:16)=xsq, ws[16:36)=csq
}

// ---------------------------------------------------------------------------
// Kernel 0b: one-shot fp32 -> bf16 conversion into workspace
// ---------------------------------------------------------------------------
__global__ __launch_bounds__(256) void cvt_bf16_kernel(const float* __restrict__ src,
                                                       unsigned short* __restrict__ dstRaw,
                                                       int n4) {
    const int i = blockIdx.x * 256 + threadIdx.x;
    if (i < n4) {
        v4f v = *(const v4f*)&src[4 * (size_t)i];
        v4bf p;
        p[0] = (__bf16)v[0]; p[1] = (__bf16)v[1];
        p[2] = (__bf16)v[2]; p[3] = (__bf16)v[3];
        *(v4bf*)((__bf16*)dstRaw + 4 * (size_t)i) = p;
    }
}

// ---------------------------------------------------------------------------
// Shared epilogue for both corr kernels: cross-wave K-split reduction + write
// dist[b,t,c] = xsq[b] + csq[c] - 2*corr
// ---------------------------------------------------------------------------
__device__ __forceinline__ void corr_epilogue(float (*red)[2][8][32],
                                              const v8f& acc0, const v8f& acc1,
                                              const float* __restrict__ sq,
                                              float* __restrict__ distOut,
                                              int t, int wave, int lane, int m, int h) {
    __syncthreads();
#pragma unroll
    for (int i = 0; i < 8; ++i) {
        red[wave][0][i][lane] = acc0[i];
        red[wave][1][i][lane] = acc1[i];
    }
    __syncthreads();
    if (wave < 2) {
        const int tile = wave;
#pragma unroll
        for (int i = 0; i < 8; ++i) {
            float ssum = 0.0f;
#pragma unroll
            for (int w = 0; w < 8; ++w) ssum += red[w][tile][i][lane];
            const int b = i + 8 * h;      // D row
            const int c = tile * 16 + m;  // D col
            if (c < C_) {
                float d = sq[b] + sq[B_ + c] - 2.0f * ssum;
                distOut[((size_t)b * T_ + t) * C_ + c] = d;
            }
        }
    }
}

// ---------------------------------------------------------------------------
// Shared WMMA compute step (reads staged bf16 tiles, 2 K-steps per wave)
// ---------------------------------------------------------------------------
__device__ __forceinline__ void corr_compute(const __bf16 (*ldsA)[KC],
                                             const __bf16 (*ldsB)[KC],
                                             v8f& acc0, v8f& acc1,
                                             int wave, int m, int h) {
#pragma unroll
    for (int kk = 0; kk < 2; ++kk) {
        const int kb = (wave + 8 * kk) * 32;
        // A lane layout (16-bit A 16x32): K = 16h..16h+15, row = m
        v8bf alo = *(const v8bf*)&ldsA[m][kb + 16 * h];
        v8bf ahi = *(const v8bf*)&ldsA[m][kb + 16 * h + 8];
        v16bf a = __builtin_shufflevector(alo, ahi,
            0, 1, 2, 3, 4, 5, 6, 7, 8, 9, 10, 11, 12, 13, 14, 15);
        // B lane layout (16-bit B 32x16): K = 16h..16h+15, col = m
        v8bf b0l = *(const v8bf*)&ldsB[m][kb + 16 * h];
        v8bf b0h = *(const v8bf*)&ldsB[m][kb + 16 * h + 8];
        v16bf b0 = __builtin_shufflevector(b0l, b0h,
            0, 1, 2, 3, 4, 5, 6, 7, 8, 9, 10, 11, 12, 13, 14, 15);
        v8bf b1l = *(const v8bf*)&ldsB[16 + m][kb + 16 * h];
        v8bf b1h = *(const v8bf*)&ldsB[16 + m][kb + 16 * h + 8];
        v16bf b1 = __builtin_shufflevector(b1l, b1h,
            0, 1, 2, 3, 4, 5, 6, 7, 8, 9, 10, 11, 12, 13, 14, 15);

        acc0 = __builtin_amdgcn_wmma_f32_16x16x32_bf16(
            false, a, false, b0, (short)0, acc0, false, false);
        acc1 = __builtin_amdgcn_wmma_f32_16x16x32_bf16(
            false, a, false, b1, (short)0, acc1, false, false);
    }
}

// ---------------------------------------------------------------------------
// Kernel 1a (preferred): corr from pre-converted bf16 sources.
// Staging is a pure b128 global->LDS copy: 9 loads + 9 ds stores per thread.
// ---------------------------------------------------------------------------
__global__ __launch_bounds__(256) void corr_bf16_kernel(const unsigned short* __restrict__ xbfRaw,
                                                        const unsigned short* __restrict__ cbfRaw,
                                                        const float* __restrict__ sq,
                                                        float* __restrict__ distOut) {
    __shared__ __bf16 ldsA[B_][KC];     // 16 KB
    __shared__ __bf16 ldsB[32][KC];     // 32 KB (rows 20..31 zero pad)
    __shared__ float  red[8][2][8][32]; // 16 KB

    const __bf16* xbf = (const __bf16*)xbfRaw;
    const __bf16* cbf = (const __bf16*)cbfRaw;

    const int t    = blockIdx.x;
    const int tid  = threadIdx.x;
    const int lane = tid & 31;
    const int wave = tid >> 5;
    const int m    = lane & 15;
    const int h    = lane >> 4;

    {
        v4bf z;
        z[0] = z[1] = z[2] = z[3] = (__bf16)0.0f;
        for (int i = tid; i < 12 * (KC / 4); i += 256) {
            int r = 20 + (i >> 7), c4 = i & 127;
            *(v4bf*)&ldsB[r][4 * c4] = z;
        }
    }

    v8f acc0 = {0.f, 0.f, 0.f, 0.f, 0.f, 0.f, 0.f, 0.f};
    v8f acc1 = {0.f, 0.f, 0.f, 0.f, 0.f, 0.f, 0.f, 0.f};

    // per-thread prefetch coordinates for A (64B per thread covers 16KB chunk)
    const int pr = tid >> 4, pg = tid & 15;

    for (int s = 0; s < T_; ++s) {
        int u = s + t;
        if (u >= T_) u -= T_;
        for (int fc = 0; fc < F_; fc += KC) {
            __syncthreads();
            // stage A: 16 rows x KC bf16 = 1024 x v8bf, 4 per thread
#pragma unroll
            for (int j = 0; j < 4; ++j) {
                int idx = tid + 256 * j;          // 0..1023
                int r = idx >> 6, g = idx & 63;   // 64 groups of 8 per row
                *(v8bf*)&ldsA[r][8 * g] =
                    *(const v8bf*)&xbf[((size_t)r * T_ + u) * F_ + fc + 8 * g];
            }
            // stage B: 20 rows x KC bf16 = 1280 x v8bf, 5 per thread
#pragma unroll
            for (int j = 0; j < 5; ++j) {
                int idx = tid + 256 * j;          // 0..1279
                int r = idx >> 6, g = idx & 63;
                *(v8bf*)&ldsB[r][8 * g] =
                    *(const v8bf*)&cbf[((size_t)r * T_ + s) * F_ + fc + 8 * g];
            }
            // prefetch next chunk's A rows (64B per thread)
            {
                int fcn = fc + KC, un = u;
                if (fcn == F_) { fcn = 0; un = u + 1; if (un >= T_) un -= T_; }
                __builtin_prefetch(&xbf[((size_t)pr * T_ + un) * F_ + fcn + 32 * pg], 0, 3);
            }
            __syncthreads();
            corr_compute(ldsA, ldsB, acc0, acc1, wave, m, h);
        }
    }
    corr_epilogue(red, acc0, acc1, sq, distOut, t, wave, lane, m, h);
}

// ---------------------------------------------------------------------------
// Kernel 1b (fallback when ws is too small): corr staging from fp32 sources
// with in-flight bf16 conversion.
// ---------------------------------------------------------------------------
__global__ __launch_bounds__(256) void corr_f32_kernel(const float* __restrict__ x,
                                                       const float* __restrict__ cent,
                                                       const float* __restrict__ sq,
                                                       float* __restrict__ distOut) {
    __shared__ __bf16 ldsA[B_][KC];
    __shared__ __bf16 ldsB[32][KC];
    __shared__ float  red[8][2][8][32];

    const int t    = blockIdx.x;
    const int tid  = threadIdx.x;
    const int lane = tid & 31;
    const int wave = tid >> 5;
    const int m    = lane & 15;
    const int h    = lane >> 4;

    {
        v4bf z;
        z[0] = z[1] = z[2] = z[3] = (__bf16)0.0f;
        for (int i = tid; i < 12 * (KC / 4); i += 256) {
            int r = 20 + (i >> 7), c4 = i & 127;
            *(v4bf*)&ldsB[r][4 * c4] = z;
        }
    }

    v8f acc0 = {0.f, 0.f, 0.f, 0.f, 0.f, 0.f, 0.f, 0.f};
    v8f acc1 = {0.f, 0.f, 0.f, 0.f, 0.f, 0.f, 0.f, 0.f};

    for (int s = 0; s < T_; ++s) {
        int u = s + t;
        if (u >= T_) u -= T_;
        for (int fc = 0; fc < F_; fc += KC) {
            __syncthreads();
#pragma unroll
            for (int j = 0; j < 8; ++j) {
                int idx = tid + 256 * j;
                int r = idx >> 7, c4 = idx & 127;
                v4f v = *(const v4f*)&x[((size_t)r * T_ + u) * F_ + fc + 4 * c4];
                v4bf p;
                p[0] = (__bf16)v[0]; p[1] = (__bf16)v[1];
                p[2] = (__bf16)v[2]; p[3] = (__bf16)v[3];
                *(v4bf*)&ldsA[r][4 * c4] = p;
            }
#pragma unroll
            for (int j = 0; j < 10; ++j) {
                int idx = tid + 256 * j;
                int r = idx >> 7, c4 = idx & 127;
                v4f v = *(const v4f*)&cent[((size_t)r * T_ + s) * F_ + fc + 4 * c4];
                v4bf p;
                p[0] = (__bf16)v[0]; p[1] = (__bf16)v[1];
                p[2] = (__bf16)v[2]; p[3] = (__bf16)v[3];
                *(v4bf*)&ldsB[r][4 * c4] = p;
            }
            __syncthreads();
            corr_compute(ldsA, ldsB, acc0, acc1, wave, m, h);
        }
    }
    corr_epilogue(red, acc0, acc1, sq, distOut, t, wave, lane, m, h);
}

// ---------------------------------------------------------------------------
// Kernel 2: per-batch L2 norm over (T,C) of dist
// ---------------------------------------------------------------------------
__global__ __launch_bounds__(256) void norm_kernel(const float* __restrict__ distOut,
                                                   float* __restrict__ normOut) {
    const int b = blockIdx.x;
    float s = 0.0f;
    for (int i = threadIdx.x; i < TC_; i += 256) {
        float v = distOut[(size_t)b * TC_ + i];
        s += v * v;
    }
    __shared__ float red[256];
    red[threadIdx.x] = s;
    __syncthreads();
    for (int off = 128; off > 0; off >>= 1) {
        if (threadIdx.x < off) red[threadIdx.x] += red[threadIdx.x + off];
        __syncthreads();
    }
    if (threadIdx.x == 0) normOut[b] = sqrtf(red[0]);
}

// ---------------------------------------------------------------------------
// Kernel 3: information einsum + fuse.  One block per t.
// dn[b,c,t] = dist_flat[b][c*T + t] / max(norm[b], eps)  (the reshape "bug"
// collapses to a flat index into the (T,C) row-major dist block).
// ---------------------------------------------------------------------------
__global__ __launch_bounds__(256) void fused_kernel(const float* __restrict__ x,
                                                    const float* __restrict__ cent,
                                                    const float* __restrict__ scaleP,
                                                    const float* __restrict__ distOut,
                                                    const float* __restrict__ norm,
                                                    float* __restrict__ fusedOut) {
    const int t   = blockIdx.x;
    const int tid = threadIdx.x;
    __shared__ float dn[C_][B_];

    for (int i = tid; i < C_ * B_; i += 256) {
        int c = i >> 4, b = i & 15;
        float rn = 1.0f / fmaxf(norm[b], 1e-12f);
        dn[c][b] = distOut[(size_t)b * TC_ + (size_t)c * T_ + t] * rn;
    }
    __syncthreads();

    const float scale = scaleP[0];
#pragma unroll 1
    for (int j = 0; j < F_ / 256; ++j) {
        const int f = tid + 256 * j;
        float cv[C_];
#pragma unroll
        for (int c = 0; c < C_; ++c)
            cv[c] = cent[((size_t)c * T_ + t) * F_ + f];
#pragma unroll
        for (int b = 0; b < B_; ++b) {
            float info = 0.0f;
#pragma unroll
            for (int c = 0; c < C_; ++c) info += dn[c][b] * cv[c];
            const size_t idx = ((size_t)b * T_ + t) * F_ + f;
            fusedOut[idx] = x[idx] + scale * info;
        }
    }
}

// ---------------------------------------------------------------------------
extern "C" void kernel_launch(void* const* d_in, const int* in_sizes, int n_in,
                              void* d_out, int out_size, void* d_ws, size_t ws_size,
                              hipStream_t stream) {
    const float* x     = (const float*)d_in[0];
    const float* cent  = (const float*)d_in[1];
    const float* scale = (const float*)d_in[2];

    float* fusedOut = (float*)d_out;             // (B,T,F)
    float* distOut  = fusedOut + FUSED_N;        // (B,T,C)

    char*  wsb  = (char*)d_ws;
    float* ws   = (float*)wsb;                   // [0:16) xsq, [16:36) csq, [36:52) norm
    unsigned short* xbf = (unsigned short*)(wsb + 256);
    unsigned short* cbf = (unsigned short*)(wsb + 256 + (size_t)2 * FUSED_N);
    const size_t need = 256 + (size_t)2 * (FUSED_N + CENT_N);

    sumsq_kernel<<<dim3(B_ + C_), dim3(256), 0, stream>>>(x, cent, ws);

    if (ws_size >= need) {
        cvt_bf16_kernel<<<dim3(FUSED_N / 4 / 256), dim3(256), 0, stream>>>(x, xbf, FUSED_N / 4);
        cvt_bf16_kernel<<<dim3(CENT_N / 4 / 256), dim3(256), 0, stream>>>(cent, cbf, CENT_N / 4);
        corr_bf16_kernel<<<dim3(T_), dim3(256), 0, stream>>>(xbf, cbf, ws, distOut);
    } else {
        corr_f32_kernel<<<dim3(T_), dim3(256), 0, stream>>>(x, cent, ws, distOut);
    }

    norm_kernel <<<dim3(B_), dim3(256), 0, stream>>>(distOut, ws + B_ + C_);
    fused_kernel<<<dim3(T_), dim3(256), 0, stream>>>(x, cent, scale, distOut,
                                                     ws + B_ + C_, fusedOut);
}
